// GAT_77953656422764
// MI455X (gfx1250) — matrix operand order repaired and checked by reading, not compiled, not run on hardware
//
#include <hip/hip_runtime.h>
#include <hip/hip_bf16.h>

// ---------- CDNA5 types ----------
typedef __attribute__((ext_vector_type(16))) _Float16 v16h;
typedef __attribute__((ext_vector_type(8)))  float    v8f;

#define NEG_SLOPE 0.2f

// ---------- helpers ----------
__device__ __forceinline__ void atomicMaxF(float* addr, float v) {
  // classic mixed-sign float max via int atomics (init must be -inf)
  if (v >= 0.0f) atomicMax((int*)addr, __float_as_int(v));
  else           atomicMin((unsigned int*)addr, __float_as_uint(v));
}

__device__ __forceinline__ float lrelu(float e) {
  return e >= 0.0f ? e : NEG_SLOPE * e;
}

// ---------- weight transpose + f32->f16 ----------
__global__ void convert_weights(const float* __restrict__ W1,
                                const float* __restrict__ W2,
                                _Float16* __restrict__ Wt1,   // [32][512]
                                _Float16* __restrict__ Wt2) { // [32][32]
  for (int idx = threadIdx.x; idx < 512 * 32; idx += blockDim.x) {
    int k = idx >> 5, n = idx & 31;
    Wt1[n * 512 + k] = (_Float16)W1[idx];
  }
  for (int idx = threadIdx.x; idx < 32 * 32; idx += blockDim.x) {
    int k = idx >> 5, n = idx & 31;
    Wt2[n * 32 + k] = (_Float16)W2[idx];
  }
}

__global__ void fill_f32(float* __restrict__ p, long long cnt, float v) {
  long long i = (long long)blockIdx.x * blockDim.x + threadIdx.x;
  long long stride = (long long)gridDim.x * blockDim.x;
  for (; i < cnt; i += stride) p[i] = v;
}

// ---------- GEMM1: h1[N,32] = x[N,512] @ W1[512,32], WMMA f16->f32 ----------
// one wave per 16-row tile; 2 accumulators cover the 32 output columns.
__global__ void gemm1_wmma(const float* __restrict__ x,
                           const _Float16* __restrict__ Wt1, // [32][512]
                           float* __restrict__ h1, int nTiles) {
  int tile = blockIdx.x * 8 + (threadIdx.x >> 5);
  if (tile >= nTiles) return;                // wave-uniform: EXEC stays all-1s
  int lane  = threadIdx.x & 31;
  int colA  = (lane & 16) ? 8 : 0;           // unused (A uses kb, contiguous K)
  int kb    = (lane & 16) ? 16 : 0;          // B-matrix K base per half-wave
  int row   = (tile << 4) + (lane & 15);
  const float*    xrow = x   + (size_t)row * 512;
  const _Float16* wc0  = Wt1 + (size_t)(lane & 15) * 512;
  const _Float16* wc1  = Wt1 + (size_t)((lane & 15) + 16) * 512;

  v8f acc0 = {}; v8f acc1 = {};
  for (int kk = 0; kk < 512; kk += 32) {
    __builtin_prefetch(xrow + kk + 32, 0, 3);  // -> global_prefetch_b8
    // A tile: f32 load, convert to f16 in registers
    v16h a;
#pragma unroll
    for (int i = 0; i < 8; i++) {
      a[i]     = (_Float16)xrow[kk + kb + i];
      a[i + 8] = (_Float16)xrow[kk + kb + 8 + i];
    }
    // B tiles: 16 contiguous halves per lane (32B load)
    v16h b0 = *(const v16h*)(wc0 + kk + kb);
    v16h b1 = *(const v16h*)(wc1 + kk + kb);
    acc0 = __builtin_amdgcn_wmma_f32_16x16x32_f16(false, a, false, b0, (short)0, acc0, false, false);
    acc1 = __builtin_amdgcn_wmma_f32_16x16x32_f16(false, a, false, b1, (short)0, acc1, false, false);
  }
  int mBase = (tile << 4) + ((lane >> 4) << 3);
  int n = lane & 15;
#pragma unroll
  for (int r = 0; r < 8; r++) {
    h1[(size_t)(mBase + r) * 32 + n]      = acc0[r];
    h1[(size_t)(mBase + r) * 32 + 16 + n] = acc1[r];
  }
}

// ---------- GEMM2: h2[N,32] = hr[N,32] @ W2[32,32] (K=32, one WMMA step) ----------
__global__ void gemm2_wmma(const _Float16* __restrict__ hr,
                           const _Float16* __restrict__ Wt2, // [32][32]
                           float* __restrict__ h2, int nTiles) {
  int tile = blockIdx.x * 8 + (threadIdx.x >> 5);
  if (tile >= nTiles) return;
  int lane = threadIdx.x & 31;
  int colA = (lane & 16) ? 8 : 0;
  int kb   = (lane & 16) ? 16 : 0;
  int row  = (tile << 4) + (lane & 15);
  const _Float16* arow = hr + (size_t)row * 32;

  v16h a;
#pragma unroll
  for (int i = 0; i < 8; i++) {
    a[i]     = arow[kb + i];
    a[i + 8] = arow[kb + 8 + i];
  }
  v16h b0 = *(const v16h*)(Wt2 + (size_t)(lane & 15) * 32 + kb);
  v16h b1 = *(const v16h*)(Wt2 + (size_t)((lane & 15) + 16) * 32 + kb);
  v8f acc0 = {}; v8f acc1 = {};
  acc0 = __builtin_amdgcn_wmma_f32_16x16x32_f16(false, a, false, b0, (short)0, acc0, false, false);
  acc1 = __builtin_amdgcn_wmma_f32_16x16x32_f16(false, a, false, b1, (short)0, acc1, false, false);

  int mBase = (tile << 4) + ((lane >> 4) << 3);
  int n = lane & 15;
#pragma unroll
  for (int r = 0; r < 8; r++) {
    h2[(size_t)(mBase + r) * 32 + n]      = acc0[r];
    h2[(size_t)(mBase + r) * 32 + 16 + n] = acc1[r];
  }
}

// ---------- attention coefficients: alpha[n,h] = <h1[n,h,:], a[h,:]> ----------
__global__ void alpha1_k(const float* __restrict__ h1,
                         const float* __restrict__ a_s, const float* __restrict__ a_d,
                         float* __restrict__ as1, float* __restrict__ ad1, int N) {
  int node = blockIdx.x * 8 + (threadIdx.x >> 5);
  if (node >= N) return;
  int lane = threadIdx.x & 31;
  float v  = h1[(size_t)node * 32 + lane];
  float ps = v * a_s[lane], pd = v * a_d[lane];
  for (int m = 1; m < 16; m <<= 1) {            // reduce each 16-lane head half
    ps += __shfl_xor(ps, m, 32);
    pd += __shfl_xor(pd, m, 32);
  }
  if ((lane & 15) == 0) {
    int h = lane >> 4;
    as1[node * 2 + h] = ps;
    ad1[node * 2 + h] = pd;
  }
}

__global__ void alpha2_k(const float* __restrict__ h2,
                         const float* __restrict__ a_s, const float* __restrict__ a_d,
                         float* __restrict__ as2, float* __restrict__ ad2, int N) {
  int node = blockIdx.x * 8 + (threadIdx.x >> 5);
  if (node >= N) return;
  int lane = threadIdx.x & 31;
  float v  = h2[(size_t)node * 32 + lane];
  float ps = v * a_s[lane], pd = v * a_d[lane];
  for (int m = 1; m < 32; m <<= 1) {
    ps += __shfl_xor(ps, m, 32);
    pd += __shfl_xor(pd, m, 32);
  }
  if (lane == 0) { as2[node] = ps; ad2[node] = pd; }
}

// ---------- edge pass 1: segment max (per head) ----------
__global__ void edge_max1(const int* __restrict__ ei, int E, int N,
                          const float* __restrict__ as1, const float* __restrict__ ad1,
                          float* __restrict__ m1) {
  int i = blockIdx.x * blockDim.x + threadIdx.x;
  int tot = E + N;
  if (i >= tot) return;
  int s, d;
  if (i < E) { s = ei[i]; d = ei[E + i]; } else { s = d = i - E; }
#pragma unroll
  for (int h = 0; h < 2; h++) {
    float e = lrelu(as1[s * 2 + h] + ad1[d * 2 + h]);
    atomicMaxF(&m1[d * 2 + h], e);
  }
}

__global__ void edge_max2(const int* __restrict__ ei, int E, int N,
                          const float* __restrict__ as2, const float* __restrict__ ad2,
                          float* __restrict__ m2) {
  int i = blockIdx.x * blockDim.x + threadIdx.x;
  int tot = E + N;
  if (i >= tot) return;
  int s, d;
  if (i < E) { s = ei[i]; d = ei[E + i]; } else { s = d = i - E; }
  float e = lrelu(as2[s] + ad2[d]);
  atomicMaxF(&m2[d], e);
}

// ---------- edge pass 2: ee=exp(e-m); denom += ee; out[dst] += ee*h[src] ----------
// one wave per edge, lane = channel (coalesced 128B gather + 128B atomic scatter)
__global__ void edge_agg1(const int* __restrict__ ei, int E, int N,
                          const float* __restrict__ as1, const float* __restrict__ ad1,
                          const float* __restrict__ m1, const float* __restrict__ h1,
                          float* __restrict__ den1, float* __restrict__ out1) {
  int i = blockIdx.x * 8 + (threadIdx.x >> 5);
  int tot = E + N;
  if (i >= tot) return;
  int lane = threadIdx.x & 31;
  int h = lane >> 4;
  int s, d;
  if (i < E) { s = ei[i]; d = ei[E + i]; } else { s = d = i - E; }
  float e  = lrelu(as1[s * 2 + h] + ad1[d * 2 + h]);
  float ee = __expf(e - m1[d * 2 + h]);
  if ((lane & 15) == 0) atomicAdd(&den1[d * 2 + h], ee);
  atomicAdd(&out1[(size_t)d * 32 + lane], ee * h1[(size_t)s * 32 + lane]);
}

__global__ void edge_agg2(const int* __restrict__ ei, int E, int N,
                          const float* __restrict__ as2, const float* __restrict__ ad2,
                          const float* __restrict__ m2, const float* __restrict__ h2,
                          float* __restrict__ den2, float* __restrict__ out2) {
  int i = blockIdx.x * 8 + (threadIdx.x >> 5);
  int tot = E + N;
  if (i >= tot) return;
  int lane = threadIdx.x & 31;
  int s, d;
  if (i < E) { s = ei[i]; d = ei[E + i]; } else { s = d = i - E; }
  float e  = lrelu(as2[s] + ad2[d]);
  float ee = __expf(e - m2[d]);
  if (lane == 0) atomicAdd(&den2[d], ee);
  atomicAdd(&out2[(size_t)d * 32 + lane], ee * h2[(size_t)s * 32 + lane]);
}

// ---------- epilogue 1: normalize, +b1, relu, cast to f16 for GEMM2 ----------
__global__ void epilogue1(const float* __restrict__ out1, const float* __restrict__ den1,
                          const float* __restrict__ b1, _Float16* __restrict__ hr, int N) {
  int node = blockIdx.x * 8 + (threadIdx.x >> 5);
  if (node >= N) return;
  int lane = threadIdx.x & 31;
  float v = out1[(size_t)node * 32 + lane] / den1[node * 2 + (lane >> 4)] + b1[lane];
  v = fmaxf(v, 0.0f);
  hr[(size_t)node * 32 + lane] = (_Float16)v;
}

// ---------- final: normalize, +b2, log_softmax over 32 classes ----------
__global__ void final_lsm(const float* __restrict__ out2, const float* __restrict__ den2,
                          const float* __restrict__ b2, float* __restrict__ out, int N) {
  int node = blockIdx.x * 8 + (threadIdx.x >> 5);
  if (node >= N) return;
  int lane = threadIdx.x & 31;
  float o = out2[(size_t)node * 32 + lane] / den2[node] + b2[lane];
  float mx = o;
  for (int m = 1; m < 32; m <<= 1) mx = fmaxf(mx, __shfl_xor(mx, m, 32));
  float l = o - mx;
  float s = __expf(l);
  for (int m = 1; m < 32; m <<= 1) s += __shfl_xor(s, m, 32);
  out[(size_t)node * 32 + lane] = l - __logf(s);
}

// ---------- host ----------
extern "C" void kernel_launch(void* const* d_in, const int* in_sizes, int n_in,
                              void* d_out, int out_size, void* d_ws, size_t ws_size,
                              hipStream_t stream) {
  const float* x      = (const float*)d_in[0];
  const int*   ei     = (const int*)d_in[1];
  const float* W1     = (const float*)d_in[2];
  const float* a_src1 = (const float*)d_in[3];
  const float* a_dst1 = (const float*)d_in[4];
  const float* b1     = (const float*)d_in[5];
  const float* W2     = (const float*)d_in[6];
  const float* a_src2 = (const float*)d_in[7];
  const float* a_dst2 = (const float*)d_in[8];
  const float* b2     = (const float*)d_in[9];
  float* out = (float*)d_out;

  const int N = in_sizes[0] / 512;
  const int E = in_sizes[1] / 2;
  const int tot = E + N;
  const int nTiles = N / 16;     // N = 100000, multiple of 16

  // workspace carve (256B aligned)
  char* base = (char*)d_ws;
  size_t off = 0;
  auto alloc = [&](size_t bytes) -> void* {
    void* p = base + off;
    off += (bytes + 255) & ~(size_t)255;
    return p;
  };
  _Float16* Wt1  = (_Float16*)alloc((size_t)512 * 32 * 2);
  _Float16* Wt2  = (_Float16*)alloc((size_t)32 * 32 * 2);
  float*    h1   = (float*)alloc((size_t)N * 32 * 4);
  float*    as1  = (float*)alloc((size_t)N * 2 * 4);
  float*    ad1  = (float*)alloc((size_t)N * 2 * 4);
  float*    m1   = (float*)alloc((size_t)N * 2 * 4);
  float*    den1 = (float*)alloc((size_t)N * 2 * 4);
  float*    out1 = (float*)alloc((size_t)N * 32 * 4);
  _Float16* hr   = (_Float16*)alloc((size_t)N * 32 * 2);
  float*    h2   = (float*)alloc((size_t)N * 32 * 4);
  float*    as2  = (float*)alloc((size_t)N * 4);
  float*    ad2  = (float*)alloc((size_t)N * 4);
  float*    m2   = (float*)alloc((size_t)N * 4);
  float*    den2 = (float*)alloc((size_t)N * 4);
  float*    out2 = (float*)alloc((size_t)N * 32 * 4);
  (void)ws_size; (void)n_in; (void)out_size;

  auto fb = [](long long c) -> int {
    long long b = (c + 255) / 256;
    return (int)(b > 4096 ? 4096 : b);
  };
  const float NEG_INF = -__builtin_inff();

  // init accumulators (every call: graph replays don't re-zero ws)
  fill_f32<<<fb(2LL * N), 256, 0, stream>>>(m1, 2LL * N, NEG_INF);
  fill_f32<<<fb(2LL * N), 256, 0, stream>>>(den1, 2LL * N, 0.0f);
  fill_f32<<<fb(32LL * N), 256, 0, stream>>>(out1, 32LL * N, 0.0f);
  fill_f32<<<fb(N), 256, 0, stream>>>(m2, N, NEG_INF);
  fill_f32<<<fb(N), 256, 0, stream>>>(den2, N, 0.0f);
  fill_f32<<<fb(32LL * N), 256, 0, stream>>>(out2, 32LL * N, 0.0f);

  convert_weights<<<1, 256, 0, stream>>>(W1, W2, Wt1, Wt2);

  // layer 1
  gemm1_wmma<<<(nTiles + 7) / 8, 256, 0, stream>>>(x, Wt1, h1, nTiles);
  alpha1_k<<<(N + 7) / 8, 256, 0, stream>>>(h1, a_src1, a_dst1, as1, ad1, N);
  edge_max1<<<(tot + 255) / 256, 256, 0, stream>>>(ei, E, N, as1, ad1, m1);
  edge_agg1<<<(tot + 7) / 8, 256, 0, stream>>>(ei, E, N, as1, ad1, m1, h1, den1, out1);
  epilogue1<<<(N + 7) / 8, 256, 0, stream>>>(out1, den1, b1, hr, N);

  // layer 2
  gemm2_wmma<<<(nTiles + 7) / 8, 256, 0, stream>>>(hr, Wt2, h2, nTiles);
  alpha2_k<<<(N + 7) / 8, 256, 0, stream>>>(h2, a_src2, a_dst2, as2, ad2, N);
  edge_max2<<<(tot + 255) / 256, 256, 0, stream>>>(ei, E, N, as2, ad2, m2);
  edge_agg2<<<(tot + 7) / 8, 256, 0, stream>>>(ei, E, N, as2, ad2, m2, h2, den2, out2);
  final_lsm<<<(N + 7) / 8, 256, 0, stream>>>(out2, den2, b2, out, N);
}
